// EquivariantGRU_19765439497063
// MI455X (gfx1250) — matrix operand order repaired and checked by reading, not compiled, not run on hardware
//
#include <hip/hip_runtime.h>
#include <math.h>

// Problem constants: SEQ,B=256, C_IN=C_OUT=64, R_IN=R_OUT=16
#define SEQ   256
#define BATCH 256
#define KDIM  1024          // C_IN * R_IN
#define NDIM  1024          // C_OUT * R_OUT
#define MTOT  (SEQ * BATCH)
#define LDSK  1032          // xproj LDS row stride (bf16 elems): 2064B = 129*16B
#define HROW  1040          // gru LDS row stride (bf16 elems): TDM pads 4dw per 256dw

typedef __attribute__((ext_vector_type(16))) __bf16 bf16x16;
typedef __attribute__((ext_vector_type(8)))  float  floatx8;
typedef __attribute__((ext_vector_type(4)))  int    v4i;
typedef __attribute__((ext_vector_type(4)))  unsigned int u32x4;
typedef __attribute__((ext_vector_type(8)))  int    i32x8;

union Frag {
    struct { uint4 lo; uint4 hi; } q;
    unsigned short us[16];
    bf16x16 v;
};

__device__ __forceinline__ unsigned short f2bf(float f) {
    union { float f; unsigned int u; } c; c.f = f;
    unsigned int u = c.u;
    u += 0x7FFFu + ((u >> 16) & 1u);   // RNE
    return (unsigned short)(u >> 16);
}
__device__ __forceinline__ float bf2f(unsigned short s) {
    union { unsigned int u; float f; } c; c.u = ((unsigned int)s) << 16;
    return c.f;
}
__device__ __forceinline__ float sigmoidf_(float x) {
    return 1.0f / (1.0f + __expf(-x));
}

#if __has_builtin(__builtin_amdgcn_global_load_async_to_lds_b128) && \
    __has_builtin(__builtin_amdgcn_s_wait_asynccnt)
#define USE_ASYNC_LDS 1
#else
#define USE_ASYNC_LDS 0
#endif

#if __has_builtin(__builtin_amdgcn_tensor_load_to_lds) && \
    __has_builtin(__builtin_amdgcn_s_wait_tensorcnt)
#define USE_TDM 1
#else
#define USE_TDM 0
#endif

// ---------------------------------------------------------------------------
__global__ void cvt_kernel(const float* __restrict__ src,
                           unsigned short* __restrict__ dst, int n) {
    int i = blockIdx.x * blockDim.x + threadIdx.x;
    if (i < n) dst[i] = f2bf(src[i]);
}

// ---------------------------------------------------------------------------
// Phase 1: hoisted x-projections for ALL THREE gates.
// Block = one 32-row A tile (staged+converted once in LDS) x 8 waves.
// Wave  = 32(M) x 32(N) x {r,u,c}: 12 accumulators; each B frag feeds 2 WMMAs.
// ---------------------------------------------------------------------------
__global__ __launch_bounds__(256)
void xproj_kernel(const float* __restrict__ x,
                  const unsigned short* __restrict__ W0, const float* __restrict__ b0, unsigned short* __restrict__ X0,
                  const unsigned short* __restrict__ W1, const float* __restrict__ b1, unsigned short* __restrict__ X1,
                  const unsigned short* __restrict__ W2, const float* __restrict__ b2, unsigned short* __restrict__ X2)
{
    __shared__ unsigned short As[32 * LDSK];   // 66 KB

    const int tid    = threadIdx.x;
    const int mtile2 = blockIdx.x >> 2;        // 0..2047 (32-row tiles)
    const int row0   = mtile2 * 32;

    // ---- Stage + convert A tile: 32x1024 f32 (contiguous 128KB) -> bf16 LDS.
    {
        const float* src = x + (size_t)row0 * KDIM;
        #pragma unroll
        for (int it = 0; it < 32; ++it) {
            const int idx4 = tid + it * 256;   // float4 index, 0..8191
            float4 f = *(const float4*)(src + idx4 * 4);
            const int pos = idx4 * 4;
            const int row = pos >> 10;
            const int k   = pos & 1023;
            union { unsigned short us[4]; unsigned long long q; } t;
            t.us[0] = f2bf(f.x); t.us[1] = f2bf(f.y);
            t.us[2] = f2bf(f.z); t.us[3] = f2bf(f.w);
            *(unsigned long long*)(As + row * LDSK + k) = t.q;
        }
    }
    __syncthreads();

    const int lane = tid & 31;
    const int wv   = tid >> 5;
    const int ngrp = (blockIdx.x & 3) * 8 + wv;   // 0..31
    const int n0   = ngrp * 32;
    const int lm   = lane & 15;
    const int half = lane >> 4;

    floatx8 acc[12];                               // [mi*6 + gate*2 + j]
    #pragma unroll
    for (int j = 0; j < 12; ++j)
        #pragma unroll
        for (int e = 0; e < 8; ++e) acc[j][e] = 0.0f;

    const unsigned short* Wg[3] = { W0, W1, W2 };
    const unsigned short* abase0 = As + lm * LDSK;
    const unsigned short* abase1 = As + (16 + lm) * LDSK;

    for (int k0 = 0; k0 < KDIM; k0 += 32) {
        Frag a0, a1;
        const unsigned short* p0 = abase0 + k0 + half * 8;
        const unsigned short* p1 = abase1 + k0 + half * 8;
        a0.q.lo = *(const uint4*)(p0); a0.q.hi = *(const uint4*)(p0 + 16);
        a1.q.lo = *(const uint4*)(p1); a1.q.hi = *(const uint4*)(p1 + 16);

        const int wofs = k0 + half * 8;
        #pragma unroll
        for (int g = 0; g < 3; ++g) {
            #pragma unroll
            for (int j = 0; j < 2; ++j) {
                const unsigned short* wp =
                    Wg[g] + (size_t)(n0 + j * 16 + lm) * KDIM + wofs;
                Frag bq;
                bq.q.lo = *(const uint4*)(wp);
                bq.q.hi = *(const uint4*)(wp + 16);
                acc[g * 2 + j] = __builtin_amdgcn_wmma_f32_16x16x32_bf16(
                    false, a0.v, false, bq.v, (short)0, acc[g * 2 + j], false, false);
                acc[6 + g * 2 + j] = __builtin_amdgcn_wmma_f32_16x16x32_bf16(
                    false, a1.v, false, bq.v, (short)0, acc[6 + g * 2 + j], false, false);
            }
        }
    }

    const float* bias[3] = { b0, b1, b2 };
    unsigned short* Xs[3] = { X0, X1, X2 };
    #pragma unroll
    for (int g = 0; g < 3; ++g) {
        #pragma unroll
        for (int j = 0; j < 2; ++j) {
            const int col = n0 + j * 16 + lm;
            const float bv = bias[g][col];
            #pragma unroll
            for (int mi = 0; mi < 2; ++mi) {
                #pragma unroll
                for (int r = 0; r < 8; ++r) {
                    const int row = row0 + mi * 16 + r + 8 * half;
                    Xs[g][(size_t)row * NDIM + col] = f2bf(acc[mi * 6 + g * 2 + j][r] + bv);
                }
            }
        }
    }
}

// ---------------------------------------------------------------------------
// Phase 2: one launch per timestep. Block = one 16-row h tile staged in LDS
// via the Tensor Data Mover (hardware-padded bank-skew layout), x 8 waves.
// ---------------------------------------------------------------------------
__global__ __launch_bounds__(256)
void gru_step_kernel(const unsigned short* __restrict__ hbf,
                     const float* __restrict__ hprev,
                     const unsigned short* __restrict__ Wrh, const float* __restrict__ brh,
                     const unsigned short* __restrict__ Wuh, const float* __restrict__ buh,
                     const unsigned short* __restrict__ Wch, const float* __restrict__ bch,
                     const unsigned short* __restrict__ Xr,
                     const unsigned short* __restrict__ Xu,
                     const unsigned short* __restrict__ Xc,
                     float* __restrict__ hseq,
                     unsigned short* __restrict__ hbf_out,
                     float* __restrict__ hlast)
{
    __shared__ unsigned short Hs[16 * HROW];   // 16 rows, stride 1040 (TDM pad)

    const int tid   = threadIdx.x;
    const int mtile = blockIdx.x >> 3;         // 0..15
    const int row0  = mtile * 16;
    const unsigned short* src = hbf + (size_t)row0 * KDIM;

#if USE_TDM
    if (tid < 32) {
        // D# group0: count=1 | lds_addr | global_addr(57b) | type=2
        const unsigned lds_base =
            (unsigned)(size_t)(__attribute__((address_space(3))) unsigned short*)Hs;
        const unsigned long long ga = (unsigned long long)(size_t)src;
        u32x4 g0;
        g0.x = 1u;                                          // count=1, user D#
        g0.y = lds_base;                                    // lds_addr (bytes)
        g0.z = (unsigned)(ga & 0xFFFFFFFFu);                // global_addr[31:0]
        g0.w = (unsigned)((ga >> 32) & 0x1FFFFFFu) | (2u << 30);  // [56:32] | type=2

        // D# group1: data_size=2B, pad: 4 dwords per 256 dwords (=> row stride 1040)
        // tensor_dim0=1024, tensor_dim1=16, tile_dim0=1024, tile_dim1=16,
        // tensor_dim0_stride=1024.
        i32x8 g1;
        g1[0] = (int)((1u << 16) | (1u << 20) | (7u << 22) | (3u << 25));
        g1[1] = (int)((1024u & 0xFFFFu) << 16);             // tensor_dim0 lo16
        g1[2] = (int)((16u & 0xFFFFu) << 16);               // dim0 hi=0 | tensor_dim1 lo16
        g1[3] = (int)(1024u << 16);                         // dim1 hi=0 | tile_dim0
        g1[4] = (int)(16u);                                 // tile_dim1=16, tile_dim2=0
        g1[5] = (int)(1024u);                               // tensor_dim0_stride lo32
        g1[6] = 0;
        g1[7] = 0;

        v4i  z4 = { 0, 0, 0, 0 };                           // 2D tile: groups 2/3 unused
        i32x8 z8 = { 0, 0, 0, 0, 0, 0, 0, 0 };
        __builtin_amdgcn_tensor_load_to_lds(g0, g1, z4, z4, z8, 0);
        __builtin_amdgcn_s_wait_tensorcnt(0);
    }
#else
    {
        #pragma unroll
        for (int it = 0; it < 8; ++it) {
            const int idx = tid + it * 256;
            const int pos = idx * 8;
            const int row = pos >> 10;
            const int k   = pos & 1023;
            const int dst = row * HROW + k + ((k & 512) ? 8 : 0);
#if USE_ASYNC_LDS
            __builtin_amdgcn_global_load_async_to_lds_b128(
                (__attribute__((address_space(1))) v4i*)(src + pos),
                (__attribute__((address_space(3))) v4i*)(Hs + dst),
                0, 0);
#else
            *(uint4*)(Hs + dst) = *(const uint4*)(src + pos);
#endif
        }
#if USE_ASYNC_LDS
        __builtin_amdgcn_s_wait_asynccnt(0);
#endif
    }
#endif
    __syncthreads();

    const int lane  = tid & 31;
    const int wv    = tid >> 5;
    const int ntile = (blockIdx.x & 7) * 8 + wv;  // 0..63
    const int n0    = ntile * 16;
    const int lm    = lane & 15;
    const int half  = lane >> 4;

    floatx8 accr, accu, accc;
    #pragma unroll
    for (int e = 0; e < 8; ++e) { accr[e] = 0.0f; accu[e] = 0.0f; accc[e] = 0.0f; }

    const unsigned short* abase = Hs + lm * HROW;
    const size_t wrow = (size_t)(n0 + lm) * KDIM;

    for (int k0 = 0; k0 < KDIM; k0 += 32) {
        const int pad = (k0 >= 512) ? 8 : 0;      // TDM pad after 512 elems/row
        Frag a;
        const unsigned short* p = abase + k0 + half * 8 + pad;
        a.q.lo = *(const uint4*)(p);
        a.q.hi = *(const uint4*)(p + 16);

        const int wofs = k0 + half * 8;
        Frag br, bu, bc;
        const unsigned short* wr = Wrh + wrow + wofs;
        const unsigned short* wu = Wuh + wrow + wofs;
        const unsigned short* wc = Wch + wrow + wofs;
        br.q.lo = *(const uint4*)(wr); br.q.hi = *(const uint4*)(wr + 16);
        bu.q.lo = *(const uint4*)(wu); bu.q.hi = *(const uint4*)(wu + 16);
        bc.q.lo = *(const uint4*)(wc); bc.q.hi = *(const uint4*)(wc + 16);

        accr = __builtin_amdgcn_wmma_f32_16x16x32_bf16(false, a.v, false, br.v, (short)0, accr, false, false);
        accu = __builtin_amdgcn_wmma_f32_16x16x32_bf16(false, a.v, false, bu.v, (short)0, accu, false, false);
        accc = __builtin_amdgcn_wmma_f32_16x16x32_bf16(false, a.v, false, bc.v, (short)0, accc, false, false);
    }

    const int o = n0 + lm;
    const float vbrh = brh[o], vbuh = buh[o], vbch = bch[o];

    #pragma unroll
    for (int r = 0; r < 8; ++r) {
        const int b = row0 + r + 8 * half;
        const size_t idx = (size_t)b * NDIM + o;
        const float hr = accr[r] + vbrh + bf2f(Xr[idx]);
        const float hu = accu[r] + vbuh + bf2f(Xu[idx]);
        const float hc = accc[r] + vbch;
        const float rg = sigmoidf_(hr);
        const float zg = sigmoidf_(hu);
        const float ng = tanhf(bf2f(Xc[idx]) + rg * hc);
        const float hn = (1.0f - zg) * ng + zg * hprev[idx];
        hseq[idx]    = hn;
        hbf_out[idx] = f2bf(hn);
        if (hlast) hlast[idx] = hn;
    }
}

// ---------------------------------------------------------------------------
// Workspace: Wb 12MB | hbf0 0.5MB | hbf1 0.5MB | Xr/Xu/Xc 3x128MB  (~397MB)
// ---------------------------------------------------------------------------
extern "C" void kernel_launch(void* const* d_in, const int* in_sizes, int n_in,
                              void* d_out, int out_size, void* d_ws, size_t ws_size,
                              hipStream_t stream)
{
    const float* x   = (const float*)d_in[0];
    const float* h0  = (const float*)d_in[1];
    const float* Wrx = (const float*)d_in[2];
    const float* brx = (const float*)d_in[3];
    const float* Wrh = (const float*)d_in[4];
    const float* brh = (const float*)d_in[5];
    const float* Wux = (const float*)d_in[6];
    const float* bux = (const float*)d_in[7];
    const float* Wuh = (const float*)d_in[8];
    const float* buh = (const float*)d_in[9];
    const float* Wcx = (const float*)d_in[10];
    const float* bcx = (const float*)d_in[11];
    const float* Wch = (const float*)d_in[12];
    const float* bch = (const float*)d_in[13];

    const size_t WSZ = (size_t)NDIM * KDIM;
    const size_t HSZ = (size_t)BATCH * NDIM;
    const size_t XSZ = (size_t)MTOT * NDIM;

    unsigned short* Wb   = (unsigned short*)d_ws;
    unsigned short* hbf0 = Wb + 6 * WSZ;
    unsigned short* hbf1 = hbf0 + HSZ;
    unsigned short* Xr   = hbf1 + HSZ;
    unsigned short* Xu   = Xr + XSZ;
    unsigned short* Xc   = Xu + XSZ;

    const int nW = (int)WSZ;
    cvt_kernel<<<(nW + 255) / 256, 256, 0, stream>>>(Wrx, Wb + 0 * WSZ, nW);
    cvt_kernel<<<(nW + 255) / 256, 256, 0, stream>>>(Wrh, Wb + 1 * WSZ, nW);
    cvt_kernel<<<(nW + 255) / 256, 256, 0, stream>>>(Wux, Wb + 2 * WSZ, nW);
    cvt_kernel<<<(nW + 255) / 256, 256, 0, stream>>>(Wuh, Wb + 3 * WSZ, nW);
    cvt_kernel<<<(nW + 255) / 256, 256, 0, stream>>>(Wcx, Wb + 4 * WSZ, nW);
    cvt_kernel<<<(nW + 255) / 256, 256, 0, stream>>>(Wch, Wb + 5 * WSZ, nW);
    cvt_kernel<<<((int)HSZ + 255) / 256, 256, 0, stream>>>(h0, hbf0, (int)HSZ);

    // Phase 1: all three x-projections in one pass (x read once from HBM).
    xproj_kernel<<<8192, 256, 0, stream>>>(x,
        Wb + 0 * WSZ, brx, Xr,
        Wb + 2 * WSZ, bux, Xu,
        Wb + 4 * WSZ, bcx, Xc);

    // Phase 2: sequential scan.
    float* hseq  = (float*)d_out;
    float* hlast = hseq + (size_t)SEQ * HSZ;
    for (int t = 0; t < SEQ; ++t) {
        const unsigned short* hin = (t & 1) ? hbf1 : hbf0;
        unsigned short* hout      = (t & 1) ? hbf0 : hbf1;
        const float* hprev = (t == 0) ? h0 : (hseq + (size_t)(t - 1) * HSZ);
        gru_step_kernel<<<128, 256, 0, stream>>>(
            hin, hprev,
            Wb + 1 * WSZ, brh,
            Wb + 3 * WSZ, buh,
            Wb + 5 * WSZ, bch,
            Xr + (size_t)t * HSZ, Xu + (size_t)t * HSZ, Xc + (size_t)t * HSZ,
            hseq + (size_t)t * HSZ, hout,
            (t == SEQ - 1) ? hlast : (float*)nullptr);
    }
}
